// ReVCosAttention_16011638080041
// MI455X (gfx1250) — matrix operand order repaired and checked by reading, not compiled, not run on hardware
//
#include <hip/hip_runtime.h>

typedef __attribute__((ext_vector_type(16))) _Float16 v16h;
typedef __attribute__((ext_vector_type(8)))  _Float16 v8h;
typedef __attribute__((ext_vector_type(8)))  float    v8f;
typedef unsigned int v4u __attribute__((ext_vector_type(4)));
typedef int v8i __attribute__((ext_vector_type(8)));
typedef int v4i __attribute__((ext_vector_type(4)));

#define L_SEQ  1024
#define DHEAD  64
#define CHUNK  64
#define NCHUNK (L_SEQ / CHUNK)
#define QS 136   // row stride (halves) for 2D=128-wide buffers (128 + 8 pad)
#define TS 72    // row stride (halves) for 64-wide buffers (64 + 8 pad)

// LDS layout (in halves)
#define OFF_Q  0                      // Qh  [64][QS]   q~ row-major
#define OFF_K  (OFF_Q  + 64  * QS)    // Kh  [64][QS]   k~ row-major (B of QK^T)
#define OFF_KT (OFF_K  + 64  * QS)    // Kt  [128][TS]  k~ transposed (B of state upd)
#define OFF_VT (OFF_KT + 128 * TS)    // Vt  [64][TS]   v transposed (B of PV, A of state upd)
#define OFF_P  (OFF_VT + 64  * TS)    // Ph  [64][TS]   masked scores, f16
#define OFF_ST (OFF_P  + 64  * TS)    // St  [64][QS]   S^T[d][k2], f16 staging of state
#define HALVES_TOTAL (OFF_ST + 64 * QS)
#define F16_BYTES (HALVES_TOTAL * 2 + 128 * 4 + 64 * 4)   // + z[128], denom[64] f32
// f32 TDM staging: double-buffered raw q/k/v chunks (64x64 f32 each)
#define STAGE_OFF  ((F16_BYTES + 255) & ~255)             // 256B aligned
#define STAGE_ELT  (CHUNK * DHEAD)                        // 4096 floats = 16 KB
#define SHMEM_BYTES (STAGE_OFF + 2 * 3 * STAGE_ELT * 4)

// ---- WMMA fragment loaders (wave32 layouts per CDNA5 ISA 7.12.2) -----------
// A (16x32 f16): lane l -> row r0+(l&15); K = k0 + (l>>4)*8 + {0..7, 16..23}
__device__ __forceinline__ v16h load_a_f16(const _Float16* base, int stride,
                                           int r0, int k0, int lane) {
  const _Float16* p = base + (r0 + (lane & 15)) * stride + k0 + ((lane >> 4) << 3);
  const v8h* pv = (const v8h*)p;         // 16-byte aligned by construction
  v8h lo = pv[0];                        // K = kb .. kb+7
  v8h hi = pv[2];                        // K = kb+16 .. kb+23
  v16h r;
#pragma unroll
  for (int i = 0; i < 8; ++i) { r[i] = lo[i]; r[i + 8] = hi[i]; }
  return r;
}

// B (32x16 f16): lane l -> col n0+(l&15); K = k0 + (l>>4)*8 + {0..7, 16..23}
__device__ __forceinline__ v16h load_b_f16(const _Float16* base, int stride,
                                           int n0, int k0, int lane) {
  const _Float16* p = base + (n0 + (lane & 15)) * stride + k0 + ((lane >> 4) << 3);
  const v8h* pv = (const v8h*)p;
  v8h lo = pv[0];
  v8h hi = pv[2];
  v16h r;
#pragma unroll
  for (int i = 0; i < 8; ++i) { r[i] = lo[i]; r[i + 8] = hi[i]; }
  return r;
}

__device__ __forceinline__ v8f wmma_f16(v16h a, v16h b, v8f c) {
  return __builtin_amdgcn_wmma_f32_16x16x32_f16(false, a, false, b,
                                                (short)0, c, false, false);
}

// ---- TDM: 2D tensor tile load, 64 rows x 64 f32, row stride 64 -------------
// D# per CDNA5 ISA 8.3/8.4 (group0: count/lds/global/type, group1: dims/tile/stride)
// 6-arg toolchain variant: (v4u g0, v8i g1, v4i g2, v4i g3, v8i g4, i32 cpol)
__device__ __forceinline__ void tdm_load_chunk(const float* gptr, unsigned lds_off) {
  unsigned long long ga = (unsigned long long)(const void*)gptr;
  v4u g0;
  g0[0] = 1u;                                   // count=1 (valid), user mode
  g0[1] = lds_off;                              // lds_addr (bytes)
  g0[2] = (unsigned)(ga & 0xffffffffu);         // global_addr[31:0]
  g0[3] = (unsigned)((ga >> 32) & 0x01ffffffu)  // global_addr[56:32]
          | (2u << 30);                         // type=2 (image)
  v8i g1;
  g1[0] = (int)(2u << 16);          // wg_mask=0, data_size=2 (4B), no pad/iter
  g1[1] = (int)(64u << 16);         // tensor_dim0[15:0]=64 at bits 63:48
  g1[2] = (int)(64u << 16);         // tensor_dim0 hi=0 | tensor_dim1[15:0]=64
  g1[3] = (int)(64u << 16);         // tensor_dim1 hi=0 | tile_dim0=64
  g1[4] = (int)(64u);               // tile_dim1=64 | tile_dim2=0
  g1[5] = (int)(64u);               // tensor_dim0_stride[31:0]=64
  g1[6] = 0;                        // stride0 hi | tensor_dim1_stride lo
  g1[7] = 0;                        // tensor_dim1_stride hi
  v4i g2 = {0, 0, 0, 0};            // 2D tensor: groups 2/3 unused
  v4i g3 = {0, 0, 0, 0};
  v8i g4 = {0, 0, 0, 0, 0, 0, 0, 0};
  __builtin_amdgcn_tensor_load_to_lds(g0, g1, g2, g3, g4, 0);
}

__device__ __forceinline__ unsigned lds_offset_of(const void* p) {
  return (unsigned)(unsigned long long)p;       // LDS aperture: addr[31:0] = offset
}

__global__ __launch_bounds__(256)
void ReVCosAttention_16011638080041_kernel(const float* __restrict__ q,
                                           const float* __restrict__ k,
                                           const float* __restrict__ v,
                                           float* __restrict__ out) {
  extern __shared__ __align__(16) char smem_raw[];
  _Float16* smem = (_Float16*)smem_raw;
  _Float16* Qh = smem + OFF_Q;
  _Float16* Kh = smem + OFF_K;
  _Float16* Kt = smem + OFF_KT;
  _Float16* Vt = smem + OFF_VT;
  _Float16* Ph = smem + OFF_P;
  _Float16* St = smem + OFF_ST;
  float* zf    = (float*)(smem + HALVES_TOTAL);
  float* denom = zf + 128;
  float* stage = (float*)(smem_raw + STAGE_OFF);  // [2][3][4096] f32

  const int tid  = threadIdx.x;
  const int lane = tid & 31;
  const int wave = tid >> 5;
  const int ncol = lane & 15;
  const int mb   = (lane >> 4) << 3;

  const size_t hbase = (size_t)blockIdx.x * L_SEQ * DHEAD;
  const float* qg = q + hbase;
  const float* kg = k + hbase;
  const float* vg = v + hbase;
  float*       og = out + hbase;

  // persistent f32 state accumulators: wave owns S^T tiles (d-block 0..3, k2-block = wave)
  v8f Sacc[4];
#pragma unroll
  for (int dt = 0; dt < 4; ++dt)
#pragma unroll
    for (int i = 0; i < 8; ++i) Sacc[dt][i] = 0.0f;

  // zero f16 state staging + z
  for (int i = tid; i < 64 * QS; i += 256) St[i] = (_Float16)0.0f;
  for (int i = tid; i < 128; i += 256) zf[i] = 0.0f;

  // waves 0/1/2 own TDM streams for q/k/v; prologue: fetch chunk 0 into buffer 0
  const float* gsrc = (wave == 0) ? qg : ((wave == 1) ? kg : vg);
  if (wave < 3 && lane == 0)
    tdm_load_chunk(gsrc, lds_offset_of(stage + (size_t)wave * STAGE_ELT));

  for (int c = 0; c < NCHUNK; ++c) {
    __syncthreads();  // prev chunk fully consumed (incl. its staging buffer)
    // prefetch chunk c+1 via TDM, then wait for chunk c's DMA (TENSORcnt in-order)
    if (wave < 3 && lane == 0) {
      if (c + 1 < NCHUNK) {
        tdm_load_chunk(gsrc + (size_t)(c + 1) * STAGE_ELT,
                       lds_offset_of(stage + (size_t)(((c + 1) & 1) * 3 + wave) * STAGE_ELT));
        __builtin_amdgcn_s_wait_tensorcnt(1);
      } else {
        __builtin_amdgcn_s_wait_tensorcnt(0);
      }
    }
    __syncthreads();  // chunk c staging visible to all waves

    const int cbase = c * CHUNK;
    const float* sq = stage + (size_t)((c & 1) * 3 + 0) * STAGE_ELT;
    const float* sk = stage + (size_t)((c & 1) * 3 + 1) * STAGE_ELT;
    const float* sv_ = stage + (size_t)((c & 1) * 3 + 2) * STAGE_ELT;

    // ---- Phase A: ReLU + cosFormer sin/cos feature map, f32 LDS -> f16 LDS --
#pragma unroll
    for (int it = 0; it < 16; ++it) {
      int idx = it * 256 + tid;         // 0..4095
      int row = idx >> 6, col = idx & 63;
      int grow = cbase + row;
      float ang = 1.5707963267948966f * (float)(grow + 1) * (1.0f / (float)L_SEQ);
      float s = __sinf(ang), cc = __cosf(ang);
      float qv = fmaxf(sq[idx], 0.0f);
      float kv = fmaxf(sk[idx], 0.0f);
      float vv = fmaxf(sv_[idx], 0.0f);
      Qh[row * QS + col]       = (_Float16)(qv * s);
      Qh[row * QS + 64 + col]  = (_Float16)(qv * cc);
      Kh[row * QS + col]       = (_Float16)(kv * s);
      Kh[row * QS + 64 + col]  = (_Float16)(kv * cc);
      Kt[col * TS + row]        = (_Float16)(kv * s);
      Kt[(col + 64) * TS + row] = (_Float16)(kv * cc);
      Vt[col * TS + row]        = (_Float16)vv;
    }
    __syncthreads();

    // ---- inter-chunk: acc = Q~ @ S_prev  (S^T staged f16 in St) ------------
    v8f acc[2];
#pragma unroll
    for (int tt = 0; tt < 2; ++tt) {
      int t = wave + tt * 8;
      int mi = t >> 2, nj = t & 3;
      v8f a = {};
#pragma unroll
      for (int ks = 0; ks < 4; ++ks)
        a = wmma_f16(load_a_f16(Qh, QS, mi * 16, ks * 32, lane),
                     load_b_f16(St, QS, nj * 16, ks * 32, lane), a);
      acc[tt] = a;
    }

    // ---- intra-chunk scores: P = mask(Q~ K~^T), stored f16 in Ph ----------
#pragma unroll
    for (int tt = 0; tt < 2; ++tt) {
      int t = wave + tt * 8;
      int mi = t >> 2, mj = t & 3;          // uniform per wave -> EXEC stays full
      if (mj <= mi) {
        v8f p = {};
#pragma unroll
        for (int ks = 0; ks < 4; ++ks)
          p = wmma_f16(load_a_f16(Qh, QS, mi * 16, ks * 32, lane),
                       load_b_f16(Kh, QS, mj * 16, ks * 32, lane), p);
#pragma unroll
        for (int r = 0; r < 8; ++r) {
          int i_loc = mi * 16 + mb + r;
          int j_loc = mj * 16 + ncol;
          float pv = (j_loc <= i_loc) ? p[r] : 0.0f;   // causal mask
          Ph[i_loc * TS + j_loc] = (_Float16)pv;
        }
      } else {
#pragma unroll
        for (int r = 0; r < 8; ++r)
          Ph[(mi * 16 + mb + r) * TS + mj * 16 + ncol] = (_Float16)0.0f;
      }
    }
    __syncthreads();

    // ---- denominator: rowsum(P) + Q~ . z_prev  (VALU; tiny) ---------------
    if (tid < 64) {
      float s = 0.0f;
#pragma unroll 8
      for (int j2 = 0; j2 < 64; ++j2) s += (float)Ph[tid * TS + j2];
      float qz = 0.0f;
#pragma unroll 8
      for (int k2 = 0; k2 < 128; ++k2) qz += (float)Qh[tid * QS + k2] * zf[k2];
      denom[tid] = fmaxf(s + qz, 1e-6f);
    }
    __syncthreads();

    // ---- P @ V, add inter part, normalize, store ---------------------------
#pragma unroll
    for (int tt = 0; tt < 2; ++tt) {
      int t = wave + tt * 8;
      int mi = t >> 2, nj = t & 3;
      v8f a = acc[tt];
#pragma unroll
      for (int ks = 0; ks < 2; ++ks)
        a = wmma_f16(load_a_f16(Ph, TS, mi * 16, ks * 32, lane),
                     load_b_f16(Vt, TS, nj * 16, ks * 32, lane), a);
#pragma unroll
      for (int r = 0; r < 8; ++r) {
        int row = mi * 16 + mb + r;
        og[(size_t)(cbase + row) * DHEAD + nj * 16 + ncol] = a[r] / denom[row];
      }
    }

    // ---- state update: S^T += V^T @ K~  -> f32 regs, stage f16 to St -------
#pragma unroll
    for (int dt = 0; dt < 4; ++dt) {
      v8f s = Sacc[dt];
#pragma unroll
      for (int ks = 0; ks < 2; ++ks)
        s = wmma_f16(load_a_f16(Vt, TS, dt * 16, ks * 32, lane),
                     load_b_f16(Kt, TS, wave * 16, ks * 32, lane), s);
      Sacc[dt] = s;
#pragma unroll
      for (int r = 0; r < 8; ++r)
        St[(dt * 16 + mb + r) * QS + wave * 16 + ncol] = (_Float16)s[r];
    }

    // ---- z update (safe: denom already consumed z behind the barrier) ------
    if (tid < 128) {
      float s = 0.0f;
#pragma unroll 8
      for (int j2 = 0; j2 < 64; ++j2) s += (float)Kt[tid * TS + j2];
      zf[tid] += s;
    }
  }
}

extern "C" void kernel_launch(void* const* d_in, const int* in_sizes, int n_in,
                              void* d_out, int out_size, void* d_ws, size_t ws_size,
                              hipStream_t stream) {
  const float* q = (const float*)d_in[0];
  const float* k = (const float*)d_in[1];
  const float* v = (const float*)d_in[2];
  float* out = (float*)d_out;
  const int n_heads = in_sizes[0] / (L_SEQ * DHEAD);  // B*H = 16
  dim3 grid(n_heads), block(256);
  hipLaunchKernelGGL(ReVCosAttention_16011638080041_kernel, grid, block,
                     (size_t)SHMEM_BYTES, stream, q, k, v, out);
}
